// HGNAM_72129680769766
// MI455X (gfx1250) — matrix operand order repaired and checked by reading, not compiled, not run on hardware
//
#include <hip/hip_runtime.h>

#define N_NODES 2048
#define F_IN    32
#define H_DIM   32
#define O_DIM   16
#define NSLICES 32
#define JSLICE  (N_NODES / NSLICES)   // 64

typedef __attribute__((ext_vector_type(16))) _Float16 v16h;
typedef __attribute__((ext_vector_type(2)))  _Float16 h2;
typedef __attribute__((ext_vector_type(8)))  float    v8f;
typedef __attribute__((ext_vector_type(4)))  float    f4;

union V16 {           // view a WMMA f16 operand as 8 packed half2's
    v16h v;
    h2   p[8];
};
union V32 {           // 32 f16 values (one lane's fs row-chunk)
    v16h v[2];
    h2   p[16];
};

// ---------------------------------------------------------------------------
// Zero-init the output (harness poisons d_out; we accumulate with atomics).
// ---------------------------------------------------------------------------
__global__ void hgnam_zero_kernel(float* __restrict__ out) {
    int t = blockIdx.x * blockDim.x + threadIdx.x;
    if (t < N_NODES * O_DIM) out[t] = 0.0f;
}

// ---------------------------------------------------------------------------
// Phase 1: f_sums[n,o] = sum_f ( relu(x[n,f]*fW1[f,:]+fb1[f,:]) @ fW2[f,:,o]
//                                + fb2[f,o] )
// 16 threads per node (one per output channel o).  Result is written f16,
// TRANSPOSED (fsT[o][n]) so phase 2 reads each lane's j-run contiguously.
// ---------------------------------------------------------------------------
__global__ void hgnam_fsums_kernel(const float* __restrict__ x,
                                   const float* __restrict__ fW1,
                                   const float* __restrict__ fb1,
                                   const float* __restrict__ fW2,
                                   const float* __restrict__ fb2,
                                   _Float16* __restrict__ fsT) {
    int t = blockIdx.x * blockDim.x + threadIdx.x;   // 0 .. N*O-1
    int n = t >> 4;
    int o = t & 15;
    float acc = 0.0f;
    for (int f = 0; f < F_IN; ++f) {
        float xv = x[n * F_IN + f];
        acc += fb2[f * O_DIM + o];
#pragma unroll
        for (int h = 0; h < H_DIM; ++h) {
            float hv = fmaxf(fmaf(xv, fW1[f * H_DIM + h], fb1[f * H_DIM + h]), 0.0f);
            acc = fmaf(hv, fW2[(f * H_DIM + h) * O_DIM + o], acc);
        }
    }
    fsT[o * N_NODES + n] = (_Float16)acc;
}

// ---------------------------------------------------------------------------
// Phase 2: pairwise message kernel as a WMMA GEMM.
//   out[i,o] = sum_j (sum_h relu(d_ij*mW1_h+mb1_h)*mW2[h,o] + mb2_o)
//              / norm_ij * fs[j,o]
// One wave per (16-row i-tile, 64-wide j-slice).  All chunk memory (dist,
// norm rows as float4, fsT as v16h) is staged into registers at the top of
// each 32-j chunk, so the WMMA body is pure packed-f16 VALU + WMMA.
// ---------------------------------------------------------------------------
__global__ __launch_bounds__(32)
void hgnam_pair_kernel(const float* __restrict__ dist,
                       const float* __restrict__ norm,
                       const _Float16* __restrict__ fsT,
                       const float* __restrict__ mW1,
                       const float* __restrict__ mb1,
                       const float* __restrict__ mW2,
                       const float* __restrict__ mb2,
                       float* __restrict__ out) {
    const int lane = threadIdx.x;        // 0..31, wave32
    const int row  = lane & 15;          // A-matrix row / B,C,D column (o)
    const int hi   = lane >> 4;          // which K-half this lane owns
    const int i    = blockIdx.x * 16 + row;
    const int jbeg = blockIdx.y * JSLICE;

    const h2 zero2 = {(_Float16)0.0f, (_Float16)0.0f};

    // Per-lane A-side weights in packed f16: 16-bit A 16x32 layout ->
    //   lanes 0-15 own K in {0..7,16..23}, lanes 16-31 own K in {8..15,24..31}
    h2 wA2[8], bA2[8];
#pragma unroll
    for (int p = 0; p < 8; ++p) {
        int k0 = 2 * p;
        int h0 = hi * 8 + ((k0 < 8) ? k0 : k0 + 8);     // pair is h0, h0+1
        wA2[p] = h2{(_Float16)mW1[h0], (_Float16)mW1[h0 + 1]};
        bA2[p] = h2{(_Float16)mb1[h0], (_Float16)mb1[h0 + 1]};
    }

    // Per-lane B-side weights: 16-bit B 32x16 layout ->
    //   lane c holds column o=c%16; lanes 0-15 K={0..7,16..23}, lanes 16-31 K={8..15,24..31}
    V16 pkB;
#pragma unroll
    for (int k = 0; k < 16; ++k)
        pkB.v[k] = (_Float16)mW2[(hi * 8 + ((k < 8) ? k : k + 8)) * O_DIM + row];

    const _Float16 mb2h = (_Float16)mb2[row];
    const h2 mb2o2 = {mb2h, mb2h};

    v8f c = {};   // f32 accumulator tile, 16x16

    const float* __restrict__ drow = dist + (size_t)i * N_NODES;
    const float* __restrict__ nrow = norm + (size_t)i * N_NODES;
    const _Float16* __restrict__ frow = fsT + (size_t)row * N_NODES;

    for (int jc = jbeg; jc < jbeg + JSLICE; jc += 32) {
        // ---- stage all chunk memory up front (pure b128 traffic) ----
        float dbuf[32], nbuf[32];
#pragma unroll
        for (int q = 0; q < 8; ++q) {
            *(f4*)(dbuf + 4 * q) = *(const f4*)(drow + jc + 4 * q);
            *(f4*)(nbuf + 4 * q) = *(const f4*)(nrow + jc + 4 * q);
        }
        V32 fsv;                                  // fs[jc..jc+31, o] in f16
        fsv.v[0] = *(const v16h*)(frow + jc);
        fsv.v[1] = *(const v16h*)(frow + jc + 16);

        // Bias-channel B: B[kk,o] = fs[jc+kk,o]*mb2_o, kk-half per lane
        V16 fsB;
#pragma unroll
        for (int p = 0; p < 8; ++p) {
            h2 sel = hi ? fsv.p[8 + p] : fsv.p[p];     // v_cndmask
            fsB.p[p] = sel * mb2o2;                    // v_pk_mul_f16
        }

        V16 aBias;
#pragma unroll
        for (int k = 0; k < 16; ++k) aBias.v[k] = (_Float16)0.0f;

#pragma unroll
        for (int jj = 0; jj < 32; ++jj) {
            const float nm = nbuf[jj];
            const float rn = __builtin_amdgcn_rcpf(nm);   // ~1ulp, rounded to
            const _Float16 rh = (_Float16)rn;             // f16 anyway
            const _Float16 dh = (_Float16)dbuf[jj];
            const h2 d2 = {dh, dh};
            const h2 r2 = {rh, rh};

            // A tile row: relu(d*w + b) / norm, all in packed f16
            V16 aV;
#pragma unroll
            for (int p = 0; p < 8; ++p) {
                h2 t = d2 * wA2[p] + bA2[p];               // v_pk_fma_f16
                t = __builtin_elementwise_max(t, zero2);   // v_pk_max_f16
                aV.p[p] = t * r2;                          // v_pk_mul_f16
            }

            // B tile: mW2(:,o) * fs[j,o] - fs splat from staged registers,
            // folds into v_pk_mul_f16 op_sel (compile-time element).
            const h2 fp = fsv.p[jj >> 1];
            const _Float16 fsh = (jj & 1) ? fp.y : fp.x;
            const h2 f2 = {fsh, fsh};
            V16 bV;
#pragma unroll
            for (int p = 0; p < 8; ++p)
                bV.p[p] = pkB.p[p] * f2;                   // v_pk_mul_f16

            c = __builtin_amdgcn_wmma_f32_16x16x32_f16(
                    false, aV.v, false, bV.v, (short)0, c, false, false);

            // Pack 1/norm into the bias-channel A matrix (K = jj).
            const int ohalf = jj >> 4;
            const int slot  = jj & 15;
            aBias.v[slot] = (hi == ohalf) ? rh : aBias.v[slot];
        }

        // Bias-channel WMMA: accumulates sum_j (1/norm_ij) * fs[j,o] * mb2_o
        c = __builtin_amdgcn_wmma_f32_16x16x32_f16(
                false, aBias.v, false, fsB.v, (short)0, c, false, false);
    }

    // C/D layout: VGPR v -> M = v + hi*8, N = row.  Combine j-slices.
#pragma unroll
    for (int v = 0; v < 8; ++v) {
        int m = v + hi * 8;
        atomicAdd(&out[(blockIdx.x * 16 + m) * O_DIM + row], c[v]);
    }
}

// ---------------------------------------------------------------------------
extern "C" void kernel_launch(void* const* d_in, const int* in_sizes, int n_in,
                              void* d_out, int out_size, void* d_ws, size_t ws_size,
                              hipStream_t stream) {
    const float* x    = (const float*)d_in[0];
    const float* dist = (const float*)d_in[1];
    const float* norm = (const float*)d_in[2];
    const float* fW1  = (const float*)d_in[3];
    const float* fb1  = (const float*)d_in[4];
    const float* fW2  = (const float*)d_in[5];
    const float* fb2  = (const float*)d_in[6];
    const float* mW1  = (const float*)d_in[7];
    const float* mb1  = (const float*)d_in[8];
    const float* mW2  = (const float*)d_in[9];
    const float* mb2  = (const float*)d_in[10];
    float* out = (float*)d_out;
    _Float16* fsT = (_Float16*)d_ws;           // O_DIM x N_NODES f16 (64 KB)

    const int nout = N_NODES * O_DIM;          // 32768
    hgnam_zero_kernel<<<(nout + 255) / 256, 256, 0, stream>>>(out);
    hgnam_fsums_kernel<<<(nout + 255) / 256, 256, 0, stream>>>(
        x, fW1, fb1, fW2, fb2, fsT);

    dim3 grid(N_NODES / 16, NSLICES);          // 128 x 32 single-wave blocks
    hgnam_pair_kernel<<<grid, 32, 0, stream>>>(
        dist, norm, fsT, mW1, mb1, mW2, mb2, out);
}
